// CSMHSA_1288490189370
// MI455X (gfx1250) — matrix operand order repaired and checked by reading, not compiled, not run on hardware
//
#include <hip/hip_runtime.h>

typedef __attribute__((ext_vector_type(16))) __bf16 v16bf;
typedef __attribute__((ext_vector_type(8)))  __bf16 v8bf;
typedef __attribute__((ext_vector_type(8)))  float  v8f;

#define S_DIM 1024
#define D_HEAD 32
#define N_HEADS 8
#define N_BATCH 8
#define C_DIM 256
#define C_HI 512

static __device__ __forceinline__ v16bf make_frag(const __bf16* p0, const __bf16* p1) {
    // elements 0..7 from p0 (16B contiguous), 8..15 from p1 (16B contiguous)
    v8bf a0 = *(const v8bf*)p0;
    v8bf a1 = *(const v8bf*)p1;
    v16bf f;
    #pragma unroll
    for (int e = 0; e < 8; ++e) { f[e] = a0[e]; f[e + 8] = a1[e]; }
    return f;
}

// ---------------------------------------------------------------------------
// WMMA projection for K and V: K/V[o,s] = sum_c W[o,c]*x_low[c,s] + b[o].
// Block: 32 o x 64 s per batch; 8 waves, one 16x16 D-tile each (K and V).
// Writes Kt [b, head, s, dd] (bf16) and V [b, head, dd, s] (bf16).
// ---------------------------------------------------------------------------
__global__ __launch_bounds__(256) void proj_kv_kernel(
    const float* __restrict__ x_low,
    const float* __restrict__ Wk, const float* __restrict__ bk,
    const float* __restrict__ Wv, const float* __restrict__ bv,
    __bf16* __restrict__ Kt, __bf16* __restrict__ Vout)
{
    __shared__ __align__(16) __bf16 xs[64][32];   // [s_local][c_local], transposed
    __shared__ __align__(16) __bf16 wks[32][32];  // [o_local][c_local]
    __shared__ __align__(16) __bf16 wvs[32][32];
    const int t    = threadIdx.x;
    const int lane = t & 31, wave = t >> 5;
    const int li   = lane & 15, hlf = lane >> 4;
    const int kb   = hlf * 8;
    const int s0   = blockIdx.x * 64;
    const int o0   = blockIdx.y * 32;
    const int b    = blockIdx.z;
    const int o_t  = (wave >> 2) * 16;   // 0 or 16
    const int s_t  = (wave & 3) * 16;    // 0,16,32,48

    v8f accK = {};
    v8f accV = {};

    for (int c0 = 0; c0 < C_DIM; c0 += 32) {
        #pragma unroll
        for (int i = 0; i < 8; ++i) {
            int idx = t + i * 256;               // 0..2047
            int cc = idx >> 6, ss = idx & 63;
            xs[ss][cc] = (__bf16)x_low[((size_t)b * C_DIM + c0 + cc) * S_DIM + s0 + ss];
        }
        #pragma unroll
        for (int i = 0; i < 4; ++i) {
            int idx = t + i * 256;               // 0..1023
            int oo = idx >> 5, cc = idx & 31;
            wks[oo][cc] = (__bf16)Wk[(size_t)(o0 + oo) * C_DIM + c0 + cc];
            wvs[oo][cc] = (__bf16)Wv[(size_t)(o0 + oo) * C_DIM + c0 + cc];
        }
        __syncthreads();
        // B fragment: lane col n=li -> s = s_t+li; k in {kb..kb+7, kb+16..kb+23}
        v16bf xf = make_frag(&xs[s_t + li][kb], &xs[s_t + li][kb + 16]);
        // A fragments: lane row m=li -> o = o0+o_t+li; k in {kb..kb+7, kb+16..kb+23}
        v16bf kfA = make_frag(&wks[o_t + li][kb], &wks[o_t + li][kb + 16]);
        v16bf vfA = make_frag(&wvs[o_t + li][kb], &wvs[o_t + li][kb + 16]);
        accK = __builtin_amdgcn_wmma_f32_16x16x32_bf16(false, kfA, false, xf, (short)0, accK, false, false);
        accV = __builtin_amdgcn_wmma_f32_16x16x32_bf16(false, vfA, false, xf, (short)0, accV, false, false);
        __syncthreads();
    }

    // D layout: m = r + 8*hlf (-> dd), n = li (-> s)
    const int s    = s0 + s_t + li;
    const int head = o0 >> 5;
    const int ddb  = o_t + 8 * hlf;
    v8bf kout;
    #pragma unroll
    for (int r = 0; r < 8; ++r) {
        int o = o0 + ddb + r;
        kout[r] = (__bf16)(accK[r] + bk[o]);
        Vout[((size_t)b * C_DIM + o) * S_DIM + s] = (__bf16)(accV[r] + bv[o]);
    }
    *(v8bf*)&Kt[(((size_t)b * N_HEADS + head) * S_DIM + s) * D_HEAD + ddb] = kout;
}

// ---------------------------------------------------------------------------
// WMMA projection for Q with fused 2x nearest-neighbor upsample of x_high.
// Writes Qt [b, head, s, dd] (bf16).
// ---------------------------------------------------------------------------
__global__ __launch_bounds__(256) void proj_q_kernel(
    const float* __restrict__ x_high,
    const float* __restrict__ Wq, const float* __restrict__ bq,
    __bf16* __restrict__ Qt)
{
    __shared__ __align__(16) __bf16 xs[64][32];
    __shared__ __align__(16) __bf16 wqs[32][32];
    const int t    = threadIdx.x;
    const int lane = t & 31, wave = t >> 5;
    const int li   = lane & 15, hlf = lane >> 4;
    const int kb   = hlf * 8;
    const int s0   = blockIdx.x * 64;
    const int o0   = blockIdx.y * 32;
    const int b    = blockIdx.z;
    const int o_t  = (wave >> 2) * 16;
    const int s_t  = (wave & 3) * 16;

    v8f acc = {};

    for (int c0 = 0; c0 < C_HI; c0 += 32) {
        #pragma unroll
        for (int i = 0; i < 8; ++i) {
            int idx = t + i * 256;
            int cc = idx >> 6, ss = idx & 63;
            int s = s0 + ss;
            int y = s >> 5, xcol = s & 31;
            int sh = (y >> 1) * 16 + (xcol >> 1);   // nearest-neighbor source
            xs[ss][cc] = (__bf16)x_high[((size_t)b * C_HI + c0 + cc) * 256 + sh];
        }
        #pragma unroll
        for (int i = 0; i < 4; ++i) {
            int idx = t + i * 256;
            int oo = idx >> 5, cc = idx & 31;
            wqs[oo][cc] = (__bf16)Wq[(size_t)(o0 + oo) * C_HI + c0 + cc];
        }
        __syncthreads();
        v16bf xf  = make_frag(&xs[s_t + li][kb], &xs[s_t + li][kb + 16]);
        v16bf qfA = make_frag(&wqs[o_t + li][kb], &wqs[o_t + li][kb + 16]);
        acc = __builtin_amdgcn_wmma_f32_16x16x32_bf16(false, qfA, false, xf, (short)0, acc, false, false);
        __syncthreads();
    }

    const int s    = s0 + s_t + li;
    const int head = o0 >> 5;
    const int ddb  = o_t + 8 * hlf;
    v8bf qout;
    #pragma unroll
    for (int r = 0; r < 8; ++r) qout[r] = (__bf16)(acc[r] + bq[o0 + ddb + r]);
    *(v8bf*)&Qt[(((size_t)b * N_HEADS + head) * S_DIM + s) * D_HEAD + ddb] = qout;
}

// ---------------------------------------------------------------------------
// Flash attention, one (b, head) per blockIdx.{z,y}; 8 waves x 16 queries.
// scores^T tile via WMMA (A=K rows, B=Q cols), online softmax in registers,
// P re-layout D->B fragment in-lane (D order = B k order), PV via WMMA (A=V rows, B=P^T).
// ---------------------------------------------------------------------------
__global__ __launch_bounds__(256) void attn_kernel(
    const __bf16* __restrict__ Qt, const __bf16* __restrict__ Kt,
    const __bf16* __restrict__ V, float* __restrict__ out)
{
    const int lane = threadIdx.x & 31;
    const int wave = threadIdx.x >> 5;
    const int li   = lane & 15;
    const int hlf  = lane >> 4;
    const int bh   = blockIdx.z * N_HEADS + blockIdx.y;
    const int iglob = blockIdx.x * 128 + wave * 16 + li;
    const int kb   = hlf * 8;

    // Q B-fragment: lane holds column n=i, d = hlf*8 + {0..7, 16..23}
    const __bf16* qp = Qt + (((size_t)bh << 10) + iglob) * D_HEAD + hlf * 8;
    v16bf qf = make_frag(qp, qp + 16);

    const __bf16* Kb = Kt + ((size_t)bh << 10) * D_HEAD;   // [j, dd]
    const __bf16* Vb = V + ((size_t)bh << 15);             // [dd, s]

    v8f acc0 = {};
    v8f acc1 = {};
    float m_i = -1e30f;
    float l_i = 0.0f;

    for (int jb = 0; jb < S_DIM; jb += 32) {
        // K A-fragments: rows j = jb+li and jb+16+li; k = kb + {0..7, 16..23}
        const __bf16* kr0 = Kb + (size_t)(jb + li) * D_HEAD + kb;
        const __bf16* kr1 = kr0 + 16 * D_HEAD;
        v16bf kf0 = make_frag(kr0, kr0 + 16);
        v16bf kf1 = make_frag(kr1, kr1 + 16);

        v8f z = {};
        // scores^T: D[m=j, n=i] = sum_d K[j,d] * Q[d,i]
        v8f s0 = __builtin_amdgcn_wmma_f32_16x16x32_bf16(false, kf0, false, qf, (short)0, z, false, false);
        v8f s1 = __builtin_amdgcn_wmma_f32_16x16x32_bf16(false, kf1, false, qf, (short)0, z, false, false);

        // online softmax over this 32-key block (per query i = lane%16)
        float bm = s0[0];
        #pragma unroll
        for (int r = 1; r < 8; ++r) bm = fmaxf(bm, s0[r]);
        #pragma unroll
        for (int r = 0; r < 8; ++r) bm = fmaxf(bm, s1[r]);
        bm = fmaxf(bm, __shfl_xor(bm, 16, 32));
        float m_new = fmaxf(m_i, bm);
        float alpha = __expf(m_i - m_new);

        float p0[8], p1[8];
        float rs = 0.0f;
        #pragma unroll
        for (int r = 0; r < 8; ++r) { p0[r] = __expf(s0[r] - m_new); rs += p0[r]; }
        #pragma unroll
        for (int r = 0; r < 8; ++r) { p1[r] = __expf(s1[r] - m_new); rs += p1[r]; }
        rs += __shfl_xor(rs, 16, 32);
        l_i = l_i * alpha + rs;
        m_i = m_new;
        #pragma unroll
        for (int r = 0; r < 8; ++r) { acc0[r] *= alpha; acc1[r] *= alpha; }

        // Re-layout P^T (D-layout) -> B-fragment (lane n=i, k = 8*hlf + {0..7, 16..23})
        v16bf pf;
        #pragma unroll
        for (int r = 0; r < 8; ++r) {
            float p0x = __shfl_xor(p0[r], 16, 32);
            float p1x = __shfl_xor(p1[r], 16, 32);
            pf[r]     = (__bf16)p0[r];
            pf[r + 8] = (__bf16)p1[r];
        }

        // V A-fragments: rows dd = li and 16+li; k = j-jb pattern
        const __bf16* vr0 = Vb + (size_t)li * S_DIM + jb + kb;
        const __bf16* vr1 = vr0 + 16 * S_DIM;
        v16bf vf0 = make_frag(vr0, vr0 + 16);
        v16bf vf1 = make_frag(vr1, vr1 + 16);

        // out[dd, i] += sum_j V[dd, j] * P[j, i]
        acc0 = __builtin_amdgcn_wmma_f32_16x16x32_bf16(false, vf0, false, pf, (short)0, acc0, false, false);
        acc1 = __builtin_amdgcn_wmma_f32_16x16x32_bf16(false, vf1, false, pf, (short)0, acc1, false, false);
    }

    float invl = 1.0f / l_i;
    #pragma unroll
    for (int r = 0; r < 8; ++r) {
        int dd = r + 8 * hlf;
        out[((size_t)bh * D_HEAD + dd) * S_DIM + iglob]      = acc0[r] * invl;
        out[((size_t)bh * D_HEAD + dd + 16) * S_DIM + iglob] = acc1[r] * invl;
    }
}

extern "C" void kernel_launch(void* const* d_in, const int* in_sizes, int n_in,
                              void* d_out, int out_size, void* d_ws, size_t ws_size,
                              hipStream_t stream) {
    const float* x_high = (const float*)d_in[0];
    const float* x_low  = (const float*)d_in[1];
    const float* Wq     = (const float*)d_in[2];
    const float* bq     = (const float*)d_in[3];
    const float* Wk     = (const float*)d_in[4];
    const float* bk     = (const float*)d_in[5];
    const float* Wv     = (const float*)d_in[6];
    const float* bv     = (const float*)d_in[7];
    float* out = (float*)d_out;

    const size_t elems = (size_t)N_BATCH * N_HEADS * S_DIM * D_HEAD;  // 2M bf16 each
    __bf16* Qt = (__bf16*)d_ws;
    __bf16* Kt = Qt + elems;
    __bf16* Vb = Kt + elems;

    dim3 blk(256);
    proj_q_kernel<<<dim3(16, 8, 8), blk, 0, stream>>>(x_high, Wq, bq, Qt);
    proj_kv_kernel<<<dim3(16, 8, 8), blk, 0, stream>>>(x_low, Wk, bk, Wv, bv, Kt, Vb);
    attn_kernel<<<dim3(8, 8, 8), blk, 0, stream>>>(Qt, Kt, Vb, out);
}
